// TFGridNetCrossAttention_47476568490619
// MI455X (gfx1250) — matrix operand order repaired and checked by reading, not compiled, not run on hardware
//
#include <hip/hip_runtime.h>
#include <hip/hip_bf16.h>

// ---------------------------------------------------------------------------
// TFGridNet cross-attention for MI455X (gfx1250, wave32, WMMA + TDM).
// Pipeline: [proj+PReLU+LN -> f16 Q/K/V staged in WMMA-friendly layouts] ->
//           [flash-attention: tensor_load_to_lds tile fills + v_wmma] ->
//           [out proj + PReLU + LN -> f32].
// ---------------------------------------------------------------------------

typedef __attribute__((ext_vector_type(8)))  _Float16 v8h;
typedef __attribute__((ext_vector_type(16))) _Float16 v16h;
typedef __attribute__((ext_vector_type(8)))  float    v8f;
typedef __attribute__((ext_vector_type(4)))  unsigned uint4v;
typedef __attribute__((ext_vector_type(8)))  unsigned uint8v;

#define BQ   4
#define HH   8
#define TT   1024
#define FN   65
#define CC   128
#define KQD  520            // E*F score-contraction length
#define SKQ  544            // padded to 17*32 for K=32 WMMA steps
#define NVD  1040           // HD*F output feature length
#define SNV  1056           // padded to 66*16 N-tiles
#define NT   66
#define LQK  552            // row stride (halfs) for Q/K tiles (bank-spread)
#define LV   40             // per-col stride (halfs) for V^T / P tiles

// ---- WMMA operand loaders (per CDNA5 ISA §7.12.2 VGPR striping) -----------
// A 16x32 f16: lanes 0-15 hold K {0..7, 16..23}; lanes 16-31 K {8..15, 24..31}
__device__ __forceinline__ v16h load_A(const _Float16* tile, int stride,
                                       int lane, int kbase) {
  const _Float16* p = tile + (lane & 15) * stride + kbase + ((lane >> 4) << 3);
  v8h lo = *(const v8h*)p;          // K = kbase + hi*8 + 0..7
  v8h hi = *(const v8h*)(p + 16);   // K = kbase + 16 + hi*8 + 0..7
  return __builtin_shufflevector(lo, hi, 0,1,2,3,4,5,6,7,8,9,10,11,12,13,14,15);
}
// B 32x16 f16: col = lane&15, lanes 0-15 K {0..7, 16..23}; lanes 16-31 K {8..15, 24..31}
__device__ __forceinline__ v16h load_B(const _Float16* tile, int stride,
                                       int lane, int kbase) {
  const _Float16* p = tile + (lane & 15) * stride + kbase + ((lane >> 4) << 3);
  v8h lo = *(const v8h*)p;
  v8h hi = *(const v8h*)(p + 16);
  return __builtin_shufflevector(lo, hi, 0,1,2,3,4,5,6,7,8,9,10,11,12,13,14,15);
}

// ---- Tensor Data Mover: contiguous 1-D memory -> LDS copy (CDNA5 §7/§8) ---
// One descriptor (count=1, type=2), data_size=4B, tile_dim0 = ndw dwords.
// Issued once per wave; completion via TENSORcnt.
__device__ __forceinline__ void tdm_load_1d(const void* lds_dst,
                                            const void* gsrc, unsigned ndw) {
  unsigned long long ga = (unsigned long long)gsrc;
  unsigned lds = (unsigned)(unsigned long long)lds_dst;  // low 32b = LDS addr
  uint4v g0;
  g0[0] = 1u;                                            // count = 1
  g0[1] = lds;                                           // lds_addr
  g0[2] = (unsigned)ga;                                  // global_addr[31:0]
  g0[3] = ((unsigned)(ga >> 32) & 0x01FFFFFFu)           // global_addr[56:32]
        | 0x80000000u;                                   // type = 2 (image)
  uint8v g1;
  g1[0] = 0x00020000u;                  // workgroup_mask=0, data_size=2 (4B)
  g1[1] = (ndw & 0xFFFFu) << 16;        // tensor_dim0[15:0]
  g1[2] = (ndw >> 16) | (1u << 16);     // tensor_dim0[31:16] | tensor_dim1=1
  g1[3] = (ndw & 0xFFFFu) << 16;        // tile_dim0 = ndw
  g1[4] = 0u;                           // tile_dim1/2 unused
  g1[5] = ndw;                          // tensor_dim0_stride[31:0]
  g1[6] = 0u;
  g1[7] = 0u;
  asm volatile("tensor_load_to_lds %0, %1" :: "s"(g0), "s"(g1) : "memory");
}

// ---------------------------------------------------------------------------
// Stage 1: y = W*x + b per (b,t); per-head PReLU; LayerNorm over (e_dim, F);
// emit f16.  vlayout==0: row-major [bh, t, padN] with padN=LQK (Q/K).
// vlayout==1: 32-row transposed panels [bh, panel, n, LV] (V for B-operand).
// ---------------------------------------------------------------------------
__global__ __launch_bounds__(256) void proj_ln_kernel(
    const float* __restrict__ X, const float* __restrict__ W,
    const float* __restrict__ bias, const float* __restrict__ gamma,
    const float* __restrict__ beta, const float* __restrict__ alpha,
    _Float16* __restrict__ out, int e_dim, int padN, int vlayout) {
  extern __shared__ char smem[];
  float* xt = (float*)smem;          // 128*65 staged input tile
  float* y  = xt + CC * FN;          // up to 128*65 pre-LN activations
  const int t = blockIdx.x, b = blockIdx.y;
  const int tid = threadIdx.x;
  const int Orows = HH * e_dim;

  for (int i = tid; i < CC * FN; i += 256) {
    int c = i / FN, f = i - c * FN;
    xt[i] = X[(((size_t)b * CC + c) * TT + t) * FN + f];
  }
  __syncthreads();

  for (int j = tid; j < Orows * FN; j += 256) {
    int o = j / FN, f = j - o * FN;
    const float* w = W + o * CC;
    float acc = bias[o];
#pragma unroll 8
    for (int c = 0; c < CC; ++c) acc += w[c] * xt[c * FN + f];
    float a = alpha[o / e_dim];
    y[j] = acc >= 0.f ? acc : a * acc;
  }
  __syncthreads();

  // One wave per head (H == 8 waves): LN over N = e_dim*F values.
  const int wave = tid >> 5, lane = tid & 31;
  const int N = e_dim * FN;
  const float* yg = y + wave * N;
  float s = 0.f, s2 = 0.f;
  for (int k = lane; k < N; k += 32) { float v = yg[k]; s += v; s2 += v * v; }
#pragma unroll
  for (int off = 16; off > 0; off >>= 1) {
    s  += __shfl_xor(s,  off, 32);
    s2 += __shfl_xor(s2, off, 32);
  }
  float mu = s / N, var = s2 / N - mu * mu;
  float rs = rsqrtf(var + 1e-5f);
  const float* g  = gamma + wave * N;
  const float* be = beta  + wave * N;
  size_t bh = (size_t)b * HH + wave;

  if (!vlayout) {
    _Float16* orow = out + (bh * TT + t) * (size_t)padN;
    for (int k = lane; k < padN; k += 32) {
      float v = 0.f;
      if (k < N) v = (yg[k] - mu) * rs * g[k] + be[k];
      orow[k] = (_Float16)v;
    }
  } else {
    int panel = t >> 5, kk = t & 31;
    _Float16* obase = out + (bh * (TT >> 5) + panel) * (size_t)(SNV * LV) + kk;
    for (int k = lane; k < padN; k += 32) {
      float v = 0.f;
      if (k < N) v = (yg[k] - mu) * rs * g[k] + be[k];
      obase[(size_t)k * LV] = (_Float16)v;
    }
  }
}

// ---------------------------------------------------------------------------
// Stage 2: flash attention.  Block = 16 q-rows of one (b,h); 8 waves.
// TDM fills K/V LDS tiles (wave 0, TENSORcnt), S via v_wmma (waves 0-1),
// online softmax (wave 0), P@V via v_wmma across 66 N-tiles on all 8 waves.
// ---------------------------------------------------------------------------
__global__ __launch_bounds__(256) void attn_kernel(
    const _Float16* __restrict__ Qg, const _Float16* __restrict__ Kg,
    const _Float16* __restrict__ Vg, _Float16* __restrict__ Og) {
  extern __shared__ char smem[];
  _Float16* sQ = (_Float16*)smem;            // 16 x LQK
  _Float16* sK = sQ + 16 * LQK;              // 32 x LQK
  _Float16* sV = sK + 32 * LQK;              // SNV x LV  (V^T panel)
  float*    sS = (float*)(sV + SNV * LV);    // 16 x 32 scores
  _Float16* sP = (_Float16*)(sS + 16 * 32);  // 16 x LV probs (f16)
  float*    sM = (float*)(sP + 16 * LV);
  float*    sL = sM + 16;
  float*    sC = sL + 16;

  const int qt = blockIdx.x, h = blockIdx.y, b = blockIdx.z;
  const int tid = threadIdx.x, wave = tid >> 5, lane = tid & 31;
  const size_t bh = (size_t)b * HH + h;
  const float scale = 0.04385290097f;        // 1/sqrt(E*F) = 1/sqrt(520)

  const _Float16* Kbh = Kg + bh * (size_t)TT * LQK;
  const _Float16* Vbh = Vg + bh * (size_t)(TT >> 5) * (SNV * LV);

  // Q tile: one TDM 1-D copy (16*LQK halfs = 4416 dwords)
  if (wave == 0) {
    tdm_load_1d(sQ, Qg + (bh * TT + (size_t)qt * 16) * LQK, 16 * LQK / 2);
  }
  if (tid < 16) { sM[tid] = -1e30f; sL[tid] = 0.f; }

  const v8f zero8 = {0.f, 0.f, 0.f, 0.f, 0.f, 0.f, 0.f, 0.f};
  v8f acc[9];
#pragma unroll
  for (int i = 0; i < 9; ++i) acc[i] = zero8;

  for (int p = 0; p < (TT >> 5); ++p) {
    if (wave == 0) {
      tdm_load_1d(sK, Kbh + (size_t)p * 32 * LQK, 32 * LQK / 2);     // 8832 dw
      tdm_load_1d(sV, Vbh + (size_t)p * (SNV * LV), SNV * LV / 2);   // 21120 dw
      __builtin_amdgcn_s_wait_tensorcnt(0);
    }
    if (p + 1 < (TT >> 5))
      __builtin_prefetch(Kbh + (size_t)(p + 1) * 32 * LQK, 0, 1);
    __syncthreads();

    // ---- S = scale * Q K^T (16x32), waves 0-1 own the two 16x16 N-tiles
    if (wave < 2) {
      v8f c = zero8;
      const _Float16* Bt = sK + (wave * 16) * LQK;
      for (int kk = 0; kk < SKQ; kk += 32) {
        v16h aOp = load_A(sQ, LQK, lane, kk);
        v16h bOp = load_B(Bt, LQK, lane, kk);
        c = __builtin_amdgcn_wmma_f32_16x16x32_f16(false, aOp, false, bOp,
                                                   (short)0, c, false, false);
      }
      int col = wave * 16 + (lane & 15);
      int rb = (lane >> 4) * 8;
#pragma unroll
      for (int r = 0; r < 8; ++r) sS[(rb + r) * 32 + col] = c[r] * scale;
    }
    __syncthreads();

    // ---- online softmax row stats (wave 0, lane == q-row)
    if (wave == 0 && lane < 16) {
      float mold = sM[lane], mnew = mold;
#pragma unroll 4
      for (int j = 0; j < 32; ++j) mnew = fmaxf(mnew, sS[lane * 32 + j]);
      float cf = __expf(mold - mnew);
      float sum = 0.f;
      for (int j = 0; j < 32; ++j) {
        float pv = __expf(sS[lane * 32 + j] - mnew);
        sum += pv;
        sP[lane * LV + j] = (_Float16)pv;
      }
      sM[lane] = mnew;
      sL[lane] = sL[lane] * cf + sum;
      sC[lane] = cf;
    }
    __syncthreads();

    // ---- rescale accumulators and add P @ V (all 8 waves, 66 N-tiles)
    int rb = (lane >> 4) * 8;
    float cf[8];
#pragma unroll
    for (int r = 0; r < 8; ++r) cf[r] = sC[rb + r];
    v16h aOp = load_A(sP, LV, lane, 0);
#pragma unroll
    for (int i = 0; i < 9; ++i) {
      int tI = wave + i * 8;
      if (tI < NT) {
        v8f c = acc[i];
#pragma unroll
        for (int r = 0; r < 8; ++r) c[r] *= cf[r];
        v16h bOp = load_B(sV + tI * 16 * LV, LV, lane, 0);
        acc[i] = __builtin_amdgcn_wmma_f32_16x16x32_f16(false, aOp, false, bOp,
                                                        (short)0, c, false, false);
      }
    }
    __syncthreads();
  }

  // ---- epilogue: normalize by l and scatter to O[b, h*16+d, q, f] (f16)
  int rb = (lane >> 4) * 8;
  float rinv[8];
#pragma unroll
  for (int r = 0; r < 8; ++r) rinv[r] = 1.0f / sL[rb + r];
  for (int i = 0; i < 9; ++i) {
    int tI = wave + i * 8;
    if (tI >= NT) break;
    int n = tI * 16 + (lane & 15);
    if (n < NVD) {
      int d = n / FN, f = n - d * FN;
      size_t base = ((size_t)b * CC + h * 16 + d) * TT + (size_t)qt * 16;
#pragma unroll
      for (int r = 0; r < 8; ++r)
        Og[(base + rb + r) * FN + f] = (_Float16)(acc[i][r] * rinv[r]);
    }
  }
}

// ---------------------------------------------------------------------------
// Stage 3: P = Wp*O + bp; scalar PReLU; LN over (C,F) per (b,t); f32 out.
// ---------------------------------------------------------------------------
__global__ __launch_bounds__(256) void out_proj_ln_kernel(
    const _Float16* __restrict__ O, const float* __restrict__ Wp,
    const float* __restrict__ bp, const float* __restrict__ ap,
    const float* __restrict__ gp, const float* __restrict__ betp,
    float* __restrict__ out) {
  extern __shared__ char smem[];
  float* ot  = (float*)smem;      // 128*65
  float* y   = ot + CC * FN;      // 128*65
  float* red = y + CC * FN;       // 18
  const int t = blockIdx.x, b = blockIdx.y;
  const int tid = threadIdx.x, wave = tid >> 5, lane = tid & 31;

  for (int i = tid; i < CC * FN; i += 256) {
    int c = i / FN, f = i - c * FN;
    ot[i] = (float)O[(((size_t)b * CC + c) * TT + t) * FN + f];
  }
  __syncthreads();

  float a0 = ap[0];
  float s = 0.f, s2 = 0.f;
  for (int j = tid; j < CC * FN; j += 256) {
    int o = j / FN, f = j - o * FN;
    const float* w = Wp + o * CC;
    float acc = bp[o];
#pragma unroll 8
    for (int c = 0; c < CC; ++c) acc += w[c] * ot[c * FN + f];
    acc = acc >= 0.f ? acc : a0 * acc;
    y[j] = acc;
    s += acc; s2 += acc * acc;
  }
#pragma unroll
  for (int off = 16; off > 0; off >>= 1) {
    s  += __shfl_xor(s,  off, 32);
    s2 += __shfl_xor(s2, off, 32);
  }
  if (lane == 0) { red[wave] = s; red[8 + wave] = s2; }
  __syncthreads();
  if (tid == 0) {
    float ts = 0.f, ts2 = 0.f;
    for (int w = 0; w < 8; ++w) { ts += red[w]; ts2 += red[8 + w]; }
    float mu = ts / (float)(CC * FN);
    float var = ts2 / (float)(CC * FN) - mu * mu;
    red[16] = mu;
    red[17] = rsqrtf(var + 1e-5f);
  }
  __syncthreads();
  float mu = red[16], rs = red[17];
  for (int j = tid; j < CC * FN; j += 256) {
    int o = j / FN, f = j - o * FN;
    out[(((size_t)b * CC + o) * TT + t) * FN + f] =
        (y[j] - mu) * rs * gp[j] + betp[j];
  }
}

// ---------------------------------------------------------------------------
extern "C" void kernel_launch(void* const* d_in, const int* in_sizes, int n_in,
                              void* d_out, int out_size, void* d_ws,
                              size_t ws_size, hipStream_t stream) {
  (void)in_sizes; (void)n_in; (void)out_size; (void)ws_size;
  const float* x    = (const float*)d_in[0];
  const float* ref  = (const float*)d_in[1];
  const float* Wq   = (const float*)d_in[2];
  const float* bq   = (const float*)d_in[3];
  const float* gq   = (const float*)d_in[4];
  const float* betq = (const float*)d_in[5];
  const float* aq   = (const float*)d_in[6];
  const float* Wk   = (const float*)d_in[7];
  const float* bk   = (const float*)d_in[8];
  const float* gk   = (const float*)d_in[9];
  const float* betk = (const float*)d_in[10];
  const float* ak   = (const float*)d_in[11];
  const float* Wv   = (const float*)d_in[12];
  const float* bv   = (const float*)d_in[13];
  const float* gv   = (const float*)d_in[14];
  const float* betv = (const float*)d_in[15];
  const float* av   = (const float*)d_in[16];
  const float* Wp   = (const float*)d_in[17];
  const float* bp   = (const float*)d_in[18];
  const float* ap   = (const float*)d_in[19];
  const float* gp   = (const float*)d_in[20];
  const float* betp = (const float*)d_in[21];

  // workspace carving (all f16; padded strides match the LDS tile layouts so
  // every TDM transfer is a contiguous 1-D copy)
  const size_t qbytes = (size_t)BQ * HH * TT * LQK * 2;          // ~36.2 MB
  const size_t kbytes = qbytes;
  const size_t vbytes = (size_t)BQ * HH * (TT >> 5) * SNV * LV * 2; // ~86.5 MB
  char* ws = (char*)d_ws;
  _Float16* Qg = (_Float16*)ws;                 ws += qbytes;
  _Float16* Kg = (_Float16*)ws;                 ws += kbytes;
  _Float16* Vg = (_Float16*)ws;                 ws += vbytes;
  _Float16* Og = (_Float16*)ws;                 // ~68.2 MB

  const dim3 pg(TT, BQ);
  const size_t shmemP = (size_t)(CC * FN + CC * FN) * 4;           // 66560 B
  proj_ln_kernel<<<pg, 256, shmemP, stream>>>(x,   Wq, bq, gq, betq, aq, Qg, 8,  LQK, 0);
  proj_ln_kernel<<<pg, 256, shmemP, stream>>>(ref, Wk, bk, gk, betk, ak, Kg, 8,  LQK, 0);
  proj_ln_kernel<<<pg, 256, shmemP, stream>>>(ref, Wv, bv, gv, betv, av, Vg, 16, SNV, 1);

  const dim3 ag(TT / 16, HH, BQ);
  const size_t shmemA = (size_t)(16 * LQK + 32 * LQK + SNV * LV) * 2  // tiles
                      + (size_t)(16 * 32) * 4                          // sS
                      + (size_t)(16 * LV) * 2                          // sP
                      + 48 * 4;                                        // stats
  attn_kernel<<<ag, 256, shmemA, stream>>>(Qg, Kg, Vg, Og);

  const dim3 fg(TT, BQ);
  const size_t shmemF = (size_t)(CC * FN * 2 + 18) * 4;
  out_proj_ln_kernel<<<fg, 256, shmemF, stream>>>(Og, Wp, bp, ap, gp, betp,
                                                  (float*)d_out);
}
